// SelfAttention_59717225284352
// MI455X (gfx1250) — compile-verified, matrix operand order checked by reading, not run on hardware
//
#include <hip/hip_runtime.h>
#include <hip/hip_bf16.h>

typedef __attribute__((ext_vector_type(16))) _Float16 v16h;
typedef __attribute__((ext_vector_type(8)))  float    v8f;
typedef __attribute__((ext_vector_type(4)))  float    v4f;
typedef __attribute__((ext_vector_type(4)))  unsigned int v4u;
typedef __attribute__((ext_vector_type(4)))  int      v4i;

#define M_TOT 16384
#define N_TOT 1024
#define K_TOT 1024
#define LDS_STRIDE 48   // 32 data + 16 pad halves: rows stay 16B-aligned, banks spread

#if __has_builtin(__builtin_amdgcn_global_load_async_to_lds_b128)
#define HAVE_ASYNC 1
typedef __attribute__((address_space(1))) v4i as1_v4i;  // global 16B vector
typedef __attribute__((address_space(3))) v4i as3_v4i;  // LDS 16B vector
#else
#define HAVE_ASYNC 0
#endif

// 16-byte global -> LDS copy: async DMA path on gfx1250, VGPR round-trip fallback
__device__ __forceinline__ void copy16_g2l(const _Float16* g, _Float16* l) {
#if HAVE_ASYNC
    __builtin_amdgcn_global_load_async_to_lds_b128((as1_v4i*)g, (as3_v4i*)l, 0, 0);
#else
    *(v4u*)l = *(const v4u*)g;
#endif
}

__device__ __forceinline__ void wait_staging() {
#if HAVE_ASYNC
#if __has_builtin(__builtin_amdgcn_s_wait_asynccnt)
    __builtin_amdgcn_s_wait_asynccnt(0);
#else
    asm volatile("s_wait_asynccnt 0x0" ::: "memory");
#endif
#endif
    __syncthreads();
}

// ---------------------------------------------------------------------------
// x (fp32) -> f16, vectorized 8 elements/thread
// ---------------------------------------------------------------------------
__global__ __launch_bounds__(256) void x_to_h(const float* __restrict__ x,
                                              _Float16* __restrict__ xh) {
    size_t i = ((size_t)blockIdx.x * 256 + threadIdx.x) * 8;
    v4f a = *(const v4f*)(x + i);
    v4f b = *(const v4f*)(x + i + 4);
    _Float16 t[8] = {(_Float16)a.x, (_Float16)a.y, (_Float16)a.z, (_Float16)a.w,
                     (_Float16)b.x, (_Float16)b.y, (_Float16)b.z, (_Float16)b.w};
    *(v4u*)(xh + i) = *(const v4u*)t;
}

// ---------------------------------------------------------------------------
// Expand quaternion weights (4,256,256) into TRANSPOSED Hamilton matrix:
// out[w][n][k] = Wfull[k][n]  (N x K, f16) so GEMM B-tiles are K-major.
// component = p^q (p=k/256, q=n/256), sign from SIGNMASK bit p*4+q
// ---------------------------------------------------------------------------
__global__ __launch_bounds__(256) void build_w(const float* __restrict__ Wq,
                                               const float* __restrict__ Wk,
                                               const float* __restrict__ Wv,
                                               const float* __restrict__ Wo,
                                               _Float16* __restrict__ out) {
    const unsigned SIGNMASK = 21392u; // bits 4,7,8,9,12,14 negative
    int idx = blockIdx.x * 256 + threadIdx.x;     // 0 .. 4*1024*1024-1
    int w   = idx >> 20;
    int rc  = idx & 1048575;
    int n = rc >> 10, k = rc & 1023;
    int p = k >> 8, a = k & 255;
    int q = n >> 8, b = n & 255;
    const float* W = (w == 0) ? Wq : (w == 1) ? Wk : (w == 2) ? Wv : Wo;
    int comp = p ^ q;
    float v = W[comp * 65536 + a * 256 + b];
    if (SIGNMASK & (1u << (p * 4 + q))) v = -v;
    out[idx] = (_Float16)v;
}

// ---------------------------------------------------------------------------
// Tiled WMMA GEMM: C(MxN) = A(MxK,f16) * Bt(NxK,f16)^T, f32 accumulate.
// Block 128x128, 256 threads (8 waves), wave tile 32x64 (2x4 WMMA C tiles).
// Double-buffered async pipeline: DMA tile t+1 into LDS while WMMAs run on t;
// one (s_wait_asynccnt + barrier) per k-step.
// ---------------------------------------------------------------------------
template <typename OT>
__global__ __launch_bounds__(256) void qgemm(const _Float16* __restrict__ A,
                                             const _Float16* __restrict__ Bt,
                                             OT* __restrict__ C) {
    __shared__ _Float16 sA[2][128][LDS_STRIDE];
    __shared__ _Float16 sB[2][128][LDS_STRIDE];

    const int tid  = threadIdx.x;
    const int lane = tid & 31;
    const int wave = tid >> 5;
    const int wm   = wave >> 1;            // 0..3  (rows of 32)
    const int wn   = wave & 1;             // 0..1  (cols of 64)
    const int m0   = blockIdx.y * 128;
    const int n0   = blockIdx.x * 128;

    const int lrow  = lane & 15;           // row/col within 16x16 tile
    const int kbase = (lane >> 4) << 3;    // 0 or 8: K-half select per WMMA layout

    v8f acc[2][4];
#pragma unroll
    for (int i = 0; i < 2; ++i)
#pragma unroll
        for (int j = 0; j < 4; ++j) acc[i][j] = (v8f)0.0f;

    const int r  = tid >> 2;               // 0..63
    const int c8 = (tid & 3) << 3;         // 0,8,16,24

    // stage one 128x32 K-major tile pair into buffer `buf`
    auto stage = [&](int buf, int kk) {
#pragma unroll
        for (int p = 0; p < 2; ++p) {
            int row = r + p * 64;
            copy16_g2l(A  + (size_t)(m0 + row) * K_TOT + kk + c8, &sA[buf][row][c8]);
            copy16_g2l(Bt + (size_t)(n0 + row) * K_TOT + kk + c8, &sB[buf][row][c8]);
        }
    };

    stage(0, 0);
    wait_staging();

    for (int kk = 0; kk < K_TOT; kk += 32) {
        const int buf = (kk >> 5) & 1;
        if (kk + 32 < K_TOT) stage(buf ^ 1, kk + 32);   // overlap DMA with WMMA

        // ---- fragments: two contiguous 16B LDS reads each (K halves) ----
        v16h af[2], bf[4];
#pragma unroll
        for (int i = 0; i < 2; ++i) {
            int row = wm * 32 + i * 16 + lrow;
            union { v16h h; v4u u[2]; } fa;
            fa.u[0] = *(const v4u*)&sA[buf][row][kbase];
            fa.u[1] = *(const v4u*)&sA[buf][row][kbase + 16];
            af[i] = fa.h;
        }
#pragma unroll
        for (int j = 0; j < 4; ++j) {
            int col = wn * 64 + j * 16 + lrow;
            union { v16h h; v4u u[2]; } fb;
            fb.u[0] = *(const v4u*)&sB[buf][col][kbase];
            fb.u[1] = *(const v4u*)&sB[buf][col][kbase + 16];
            bf[j] = fb.h;
        }

#pragma unroll
        for (int i = 0; i < 2; ++i)
#pragma unroll
            for (int j = 0; j < 4; ++j)
                acc[i][j] = __builtin_amdgcn_wmma_f32_16x16x32_f16(
                    false, af[i], false, bf[j], (short)0, acc[i][j], false, false);

        wait_staging();   // next tile's DMA complete + all waves done reading buf
    }

    // ---- epilogue: C layout = lane holds col N=lane%16, VGPR v row = v + 8*(lane/16)
    const int rsel = (lane >> 4) << 3;
    const int ccol = n0 + wn * 64 + lrow;
#pragma unroll
    for (int i = 0; i < 2; ++i)
#pragma unroll
        for (int j = 0; j < 4; ++j)
#pragma unroll
            for (int v = 0; v < 8; ++v) {
                int row = m0 + wm * 32 + i * 16 + rsel + v;
                C[(size_t)row * N_TOT + ccol + j * 16] = (OT)acc[i][j][v];
            }
}

// ---------------------------------------------------------------------------
// Fused elementwise: logits=0.06*ham(Q,K); att=logits/sqrt(|logits|^2+1e-8);
// t = ham(att, V) (f16). th may alias Qh (same-thread same-index RAW only).
// ---------------------------------------------------------------------------
__global__ __launch_bounds__(256) void elem1(const _Float16* Qh,
                                             const _Float16* __restrict__ Kh,
                                             const _Float16* __restrict__ Vh,
                                             float* __restrict__ att,
                                             _Float16* th) {
    int tid = blockIdx.x * 256 + threadIdx.x;  // 16384*256 total
    int row = tid >> 8;
    int j   = tid & 255;
    size_t base = (size_t)row * 1024 + j;

    float a = (float)Qh[base], b = (float)Qh[base + 256],
          c = (float)Qh[base + 512], d = (float)Qh[base + 768];
    float e = (float)Kh[base], f = (float)Kh[base + 256],
          g = (float)Kh[base + 512], h = (float)Kh[base + 768];

    float lr = a * e - b * f - c * g - d * h;
    float li = a * f + b * e + c * h - d * g;
    float lj = a * g - b * h + c * e + d * f;
    float lk = a * h + b * g - c * f + d * e;
    lr *= 0.06f; li *= 0.06f; lj *= 0.06f; lk *= 0.06f;

    float inv = 1.0f / sqrtf(lr * lr + li * li + lj * lj + lk * lk + 1e-8f);
    float ar = lr * inv, ai = li * inv, aj = lj * inv, ak = lk * inv;
    att[base] = ar; att[base + 256] = ai; att[base + 512] = aj; att[base + 768] = ak;

    float ve = (float)Vh[base], vf = (float)Vh[base + 256],
          vg = (float)Vh[base + 512], vh = (float)Vh[base + 768];
    float tr = ar * ve - ai * vf - aj * vg - ak * vh;
    float ti = ar * vf + ai * ve + aj * vh - ak * vg;
    float tj = ar * vg - ai * vh + aj * ve + ak * vf;
    float tk = ar * vh + ai * vg - aj * vf + ak * ve;
    th[base]       = (_Float16)tr;
    th[base + 256] = (_Float16)ti;
    th[base + 512] = (_Float16)tj;
    th[base + 768] = (_Float16)tk;
}

// ---------------------------------------------------------------------------
extern "C" void kernel_launch(void* const* d_in, const int* in_sizes, int n_in,
                              void* d_out, int out_size, void* d_ws, size_t ws_size,
                              hipStream_t stream) {
    const float* x  = (const float*)d_in[0];
    const float* Wq = (const float*)d_in[1];
    const float* Wk = (const float*)d_in[2];
    const float* Wv = (const float*)d_in[3];
    const float* Wo = (const float*)d_in[4];

    float* y_out   = (float*)d_out;                 // 16384*1024
    float* att_out = y_out + (size_t)M_TOT * N_TOT; // second tuple element

    // workspace layout (total 136 MiB)
    _Float16* Wfull = (_Float16*)d_ws;                          // 4 * 1Mi halves = 8 MiB
    _Float16* xh    = (_Float16*)((char*)d_ws + (size_t)8 * 1024 * 1024);  // 32 MiB
    _Float16* Qh    = xh + (size_t)M_TOT * K_TOT;               // 32 MiB
    _Float16* Kh    = Qh + (size_t)M_TOT * N_TOT;               // 32 MiB
    _Float16* Vh    = Kh + (size_t)M_TOT * N_TOT;               // 32 MiB
    _Float16* th    = Qh;  // alias: Q dead after elem1 reads it

    x_to_h<<<(size_t)M_TOT * K_TOT / (256 * 8), 256, 0, stream>>>(x, xh);
    build_w<<<4 * 1024 * 1024 / 256, 256, 0, stream>>>(Wq, Wk, Wv, Wo, Wfull);

    dim3 grid(N_TOT / 128, M_TOT / 128), blk(256);
    const size_t WSTRIDE = (size_t)N_TOT * K_TOT;
    qgemm<_Float16><<<grid, blk, 0, stream>>>(xh, Wfull + 0 * WSTRIDE, Qh);
    qgemm<_Float16><<<grid, blk, 0, stream>>>(xh, Wfull + 1 * WSTRIDE, Kh);
    qgemm<_Float16><<<grid, blk, 0, stream>>>(xh, Wfull + 2 * WSTRIDE, Vh);

    elem1<<<M_TOT, 256, 0, stream>>>(Qh, Kh, Vh, att_out, th);

    qgemm<float><<<grid, blk, 0, stream>>>(th, Wfull + 3 * WSTRIDE, y_out);
}
